// XdiffMultiHeadAttention_60404420051700
// MI455X (gfx1250) — matrix operand order repaired and checked by reading, not compiled, not run on hardware
//
#include <hip/hip_runtime.h>
#include <hip/hip_bf16.h>
#include <math.h>

typedef __attribute__((ext_vector_type(16))) _Float16 v16h;
typedef __attribute__((ext_vector_type(8)))  _Float16 v8h;
typedef __attribute__((ext_vector_type(2)))  _Float16 v2h;
typedef __attribute__((ext_vector_type(8)))  float    v8f;

constexpr int cB  = 4;
constexpr int cS  = 1024;
constexpr int cDM = 512;
constexpr int cDA = 512;
constexpr int cH  = 8;
constexpr int cF  = 16;
constexpr int cD  = 64;      // per-head depth
constexpr int cHF = cH * cF; // 128

// Convert 8 consecutive f32 (16B-aligned) to v8h via two float4 loads.
__device__ __forceinline__ v8h cvt8(const float* __restrict__ p) {
    float4 a = ((const float4*)p)[0];
    float4 b = ((const float4*)p)[1];
    v8h r;
    r[0] = (_Float16)a.x; r[1] = (_Float16)a.y; r[2] = (_Float16)a.z; r[3] = (_Float16)a.w;
    r[4] = (_Float16)b.x; r[5] = (_Float16)b.y; r[6] = (_Float16)b.z; r[7] = (_Float16)b.w;
    return r;
}
__device__ __forceinline__ v16h cat16(v8h lo, v8h hi) {
    return __builtin_shufflevector(lo, hi, 0,1,2,3,4,5,6,7,8,9,10,11,12,13,14,15);
}
// A-fragment (16-bit 16x32) from row-major LDS tile with row stride `stride`
// halves 0..7 at K=8*hi, halves 8..15 at K=16+8*hi  -> two b128 LDS loads
__device__ __forceinline__ v16h afrag(const _Float16* __restrict__ base) {
    v8h lo = *(const v8h*)(base);
    v8h hi = *(const v8h*)(base + 16);
    return cat16(lo, hi);
}

// ---------------------------------------------------------------------------
// Tiled GEMM with bias: C[M,N] = A[M,K] @ W[K,N] + bias[N]
// f32 in/out, v_wmma_f32_16x16x32_f16. Block = 256 thr = 8 waves,
// macro-tile 16(M) x 128(N), one 16x16 WMMA tile per wave.
// A staged row-major (16x32), B staged K-transposed (128 x 32) so both
// fragments are contiguous b128 LDS loads.
// ---------------------------------------------------------------------------
__global__ __launch_bounds__(256)
void gemm_bias_f16wmma(const float* __restrict__ A, const float* __restrict__ W,
                       const float* __restrict__ bias, float* __restrict__ C,
                       int M, int N, int K)
{
    __shared__ _Float16 Asub[16 * 32];     // [m][k]
    __shared__ _Float16 Bsubt[128 * 32];   // [n][k]  (K-transposed)

    const int tid  = threadIdx.x;
    const int lane = tid & 31;
    const int wave = tid >> 5;
    const int hi   = lane >> 4;
    const int ln   = lane & 15;
    const int m0   = blockIdx.y * 16;
    const int nblk = blockIdx.x * 128;

    // B staging mapping: col = tid&127 (N), grp = tid>>7 selects K rows [grp*16, +16)
    const int bcol = tid & 127;
    const int bk0r = (tid >> 7) * 16;

    v8f acc = {};
    for (int kk = 0; kk < K; kk += 32) {
        // Stage A tile 16x32: 2 consecutive elems/thread -> 1 ds_store_b32
        {
            int r = tid >> 4, c = (tid * 2) & 31;
            float2 t = *(const float2*)&A[(size_t)(m0 + r) * K + kk + c];
            v2h p; p[0] = (_Float16)t.x; p[1] = (_Float16)t.y;
            *(v2h*)&Asub[r * 32 + c] = p;
        }
        // Stage B tile (transposed): each thread owns a 16-long K-run of one column
        {
            v8h t0, t1;
            #pragma unroll
            for (int i = 0; i < 8; ++i) {
                t0[i] = (_Float16)W[(size_t)(kk + bk0r + i)     * N + nblk + bcol];
                t1[i] = (_Float16)W[(size_t)(kk + bk0r + 8 + i) * N + nblk + bcol];
            }
            *(v8h*)&Bsubt[bcol * 32 + bk0r]     = t0;   // ds_store_b128
            *(v8h*)&Bsubt[bcol * 32 + bk0r + 8] = t1;   // ds_store_b128
        }
        __syncthreads();

        v16h a = afrag(&Asub[ln * 32 + hi * 8]);                       // 2x ds_load_b128
        v16h b = afrag(&Bsubt[(wave * 16 + ln) * 32 + hi * 8]);         // 2x ds_load_b128
        acc = __builtin_amdgcn_wmma_f32_16x16x32_f16(false, a, false, b,
                                                     (short)0, acc, false, false);
        __syncthreads();
    }

    #pragma unroll
    for (int r = 0; r < 8; ++r) {
        int m = m0 + r + hi * 8;
        int n = nblk + wave * 16 + ln;
        C[(size_t)m * N + n] = acc[r] + bias[n];
    }
}

// ---------------------------------------------------------------------------
// Fused xdiff attention. One block (4 waves) per (b, h, 16-row q-tile).
// Wave w owns k-range [w*256, w*256+256).
//  P1: logits = QK^T/8 + 0.5*(xdiff . xh) -> f16 LDS strip (32 KB), rowmax
//  P2: p = exp(l-max); attn@V WMMA reads p straight from LDS; O via ds_add_f32
//  P3: single normalized coalesced attn write + O scatter to [B,S,DA]
// ---------------------------------------------------------------------------
__global__ __launch_bounds__(128)
void xdiff_attn_kernel(const float* __restrict__ qp, const float* __restrict__ kp,
                       const float* __restrict__ vp, const float* __restrict__ xw,
                       const float* __restrict__ xdiff,
                       float* __restrict__ attn, float* __restrict__ ocat)
{
    __shared__ _Float16 lg[16 * 1024];   // logits -> p strip (f16, 32 KB)
    __shared__ float    xh[16 * 16];
    __shared__ float    ost[16 * 64];
    __shared__ float    stat[4][16];
    __shared__ float    invs[16];

    const int tid  = threadIdx.x;
    const int lane = tid & 31;
    const int wave = tid >> 5;
    const int hi   = lane >> 4;
    const int ln   = lane & 15;
    const int q0   = blockIdx.x * 16;
    const int h    = blockIdx.y;
    const int b    = blockIdx.z;

    for (int e = tid; e < 256; e += 128) {
        int qr = e >> 4, f = e & 15;
        xh[e] = xw[(size_t)(b * cS + q0 + qr) * cHF + h * cF + f];
    }
    for (int e = tid; e < 1024; e += 128) ost[e] = 0.0f;

    // Q fragments: halves 0..15 at d=16*hi+j (per chunk)
    const float* qbase = &qp[(size_t)(b * cS + q0 + ln) * cDA + h * cD];
    v16h aq[2];
    #pragma unroll
    for (int c = 0; c < 2; ++c)
        aq[c] = cat16(cvt8(qbase + c * 32 + hi * 16),
                      cvt8(qbase + c * 32 + hi * 16 + 8));
    __syncthreads();

    const int kw0 = wave * 256;
    float mx[8], sm[8];
    #pragma unroll
    for (int r = 0; r < 8; ++r) { mx[r] = -1e30f; sm[r] = 0.0f; }

    // ---- Phase 1 -----------------------------------------------------------
    for (int kt = kw0; kt < kw0 + 256; kt += 16) {
        // B-frags: 16 consecutive d per lane starting at 16*hi (global b128 loads)
        const float* kbase = &kp[(size_t)(b * cS + kt + ln) * cDA + h * cD + hi * 16];
        v16h bk0 = cat16(cvt8(kbase),      cvt8(kbase + 8));
        v16h bk1 = cat16(cvt8(kbase + 32), cvt8(kbase + 40));

        v8f acc = {};
        acc = __builtin_amdgcn_wmma_f32_16x16x32_f16(false, aq[0], false, bk0,
                                                     (short)0, acc, false, false);
        acc = __builtin_amdgcn_wmma_f32_16x16x32_f16(false, aq[1], false, bk1,
                                                     (short)0, acc, false, false);
        #pragma unroll
        for (int r = 0; r < 8; ++r) {
            int qr = r + 8 * hi;
            const float4* xd =
                (const float4*)&xdiff[((size_t)(b * cS + q0 + qr) * cS + kt + ln) * cF];
            if (kt + 16 < kw0 + 256)
                __builtin_prefetch(((const char*)xd) + (size_t)16 * cF * 4, 0, 1);
            float s = 0.0f;
            #pragma unroll
            for (int v4 = 0; v4 < 4; ++v4) {
                float4 t = xd[v4];
                s += t.x * xh[qr * 16 + v4 * 4 + 0];
                s += t.y * xh[qr * 16 + v4 * 4 + 1];
                s += t.z * xh[qr * 16 + v4 * 4 + 2];
                s += t.w * xh[qr * 16 + v4 * 4 + 3];
            }
            float l = 0.125f * acc[r] + 0.5f * s;   // 1/sqrt(64), 2/sqrt(16)
            mx[r] = fmaxf(mx[r], l);
            lg[qr * 1024 + kt + ln] = (_Float16)l;
        }
    }
    #pragma unroll
    for (int m = 1; m < 16; m <<= 1)
        #pragma unroll
        for (int r = 0; r < 8; ++r)
            mx[r] = fmaxf(mx[r], __shfl_xor(mx[r], m, 32));
    if (ln == 0)
        #pragma unroll
        for (int r = 0; r < 8; ++r) stat[wave][8 * hi + r] = mx[r];
    __syncthreads();
    #pragma unroll
    for (int r = 0; r < 8; ++r) {
        float m = fmaxf(fmaxf(stat[0][8 * hi + r], stat[1][8 * hi + r]),
                        fmaxf(stat[2][8 * hi + r], stat[3][8 * hi + r]));
        mx[r] = m;
    }
    __syncthreads();   // stat reused for sums

    // ---- Phase 2 -----------------------------------------------------------
    v8f o[4] = { v8f{}, v8f{}, v8f{}, v8f{} };
    for (int kt = kw0; kt < kw0 + 256; kt += 32) {
        #pragma unroll
        for (int sub = 0; sub < 2; ++sub) {
            #pragma unroll
            for (int r = 0; r < 8; ++r) {
                int qr = r + 8 * hi;
                int idx = qr * 1024 + kt + sub * 16 + ln;
                float p = __expf((float)lg[idx] - mx[r]);
                sm[r] += p;
                lg[idx] = (_Float16)p;
            }
        }
        // A-frag of attn@V straight from the strip (two b128 LDS loads)
        v16h ap = *(const v16h*)&lg[ln * 1024 + kt + hi * 16];
        #pragma unroll
        for (int t = 0; t < 4; ++t) {
            const float* vbase = &vp[(size_t)(b * cS + kt + hi * 16) * cDA + h * cD + t * 16 + ln];
            // B-frag halves j -> K=j+16*hi at fixed N=ln: rows stride cDA
            v16h bv;
            #pragma unroll
            for (int j = 0; j < 16; ++j)
                bv[j] = (_Float16)vbase[(size_t)j * cDA];
            o[t] = __builtin_amdgcn_wmma_f32_16x16x32_f16(false, ap, false, bv,
                                                          (short)0, o[t], false, false);
        }
    }
    #pragma unroll
    for (int m = 1; m < 16; m <<= 1)
        #pragma unroll
        for (int r = 0; r < 8; ++r)
            sm[r] += __shfl_xor(sm[r], m, 32);
    if (ln == 0)
        #pragma unroll
        for (int r = 0; r < 8; ++r) stat[wave][8 * hi + r] = sm[r];
    __syncthreads();
    float inv[8];
    #pragma unroll
    for (int r = 0; r < 8; ++r) {
        float s = stat[0][8 * hi + r] + stat[1][8 * hi + r] +
                  stat[2][8 * hi + r] + stat[3][8 * hi + r];
        inv[r] = 1.0f / s;
    }
    if (wave == 0 && ln == 0)
        #pragma unroll
        for (int r = 0; r < 8; ++r) invs[8 * hi + r] = inv[r];

    #pragma unroll
    for (int t = 0; t < 4; ++t)
        #pragma unroll
        for (int r = 0; r < 8; ++r)
            atomicAdd(&ost[(r + 8 * hi) * 64 + t * 16 + ln], o[t][r]);  // ds_add_f32
    __syncthreads();

    // ---- Phase 3 -----------------------------------------------------------
    float* arow = attn + ((size_t)(b * cH + h) * cS + q0) * cS;
    for (int e = tid; e < 16 * 1024; e += 128) {
        int qr = e >> 10;
        arow[(size_t)qr * cS + (e & 1023)] = (float)lg[e] * invs[qr];
    }
    for (int e = tid; e < 1024; e += 128) {
        int qr = e >> 6, d = e & 63;
        ocat[(size_t)(b * cS + q0 + qr) * cDA + h * cD + d] = ost[e] * invs[qr];
    }
}

// ---------------------------------------------------------------------------
extern "C" void kernel_launch(void* const* d_in, const int* in_sizes, int n_in,
                              void* d_out, int out_size, void* d_ws, size_t ws_size,
                              hipStream_t stream)
{
    const float* q   = (const float*)d_in[0];
    const float* k   = (const float*)d_in[1];
    const float* v   = (const float*)d_in[2];
    const float* xd  = (const float*)d_in[3];
    const float* Wq  = (const float*)d_in[4];
    const float* bq  = (const float*)d_in[5];
    const float* Wk  = (const float*)d_in[6];
    const float* bk  = (const float*)d_in[7];
    const float* Wv  = (const float*)d_in[8];
    const float* bv  = (const float*)d_in[9];
    const float* Wx  = (const float*)d_in[10];
    const float* bx  = (const float*)d_in[11];
    const float* Wo  = (const float*)d_in[12];
    const float* bo  = (const float*)d_in[13];

    const int M = cB * cS;               // 4096
    float* ws = (float*)d_ws;
    float* qp = ws;                                  // M*DA
    float* kp = qp + (size_t)M * cDA;
    float* vp = kp + (size_t)M * cDA;
    float* xw = vp + (size_t)M * cDA;                // M*HF
    float* oc = xw + (size_t)M * cHF;                // M*DA

    float* out  = (float*)d_out;                           // [B,S,DM]
    float* attn = out + (size_t)cB * cS * cDM;             // [B,H,S,S]

    dim3 blk(256);
    gemm_bias_f16wmma<<<dim3(cDA / 128, M / 16), blk, 0, stream>>>(q,  Wq, bq, qp, M, cDA, cDM);
    gemm_bias_f16wmma<<<dim3(cDA / 128, M / 16), blk, 0, stream>>>(k,  Wk, bk, kp, M, cDA, cDM);
    gemm_bias_f16wmma<<<dim3(cDA / 128, M / 16), blk, 0, stream>>>(v,  Wv, bv, vp, M, cDA, cDM);
    gemm_bias_f16wmma<<<dim3(cHF / 128, M / 16), blk, 0, stream>>>(qp, Wx, bx, xw, M, cHF, cDA);

    xdiff_attn_kernel<<<dim3(cS / 16, cH, cB), dim3(128), 0, stream>>>(qp, kp, vp, xw, xd, attn, oc);

    gemm_bias_f16wmma<<<dim3(cDM / 128, M / 16), blk, 0, stream>>>(oc, Wo, bo, out, M, cDM, cDA);
}
